// CausalSelfAttention_15685220565369
// MI455X (gfx1250) — matrix operand order repaired and checked by reading, compiled, not run on hardware
//
#include <hip/hip_runtime.h>

// ---------------------------------------------------------------------------
// CDNA5 (gfx1250) causal self-attention:
//   qkv = x @ W + b  -> split into Q,K,V (bf16, [B,H,T,d] layout, in d_ws)
//   Q is pre-scaled by sqrt(d)=8 (exact, power of two) at projection time.
//   flash-attention with causal mask (reference multiplies by sqrt(d)).
// All matmuls through v_wmma_f32_16x16x32_bf16 (wave32).
// ---------------------------------------------------------------------------

typedef __attribute__((ext_vector_type(16))) __bf16 v16bf;
typedef __attribute__((ext_vector_type(8)))  float  v8f;
typedef int v4i __attribute__((vector_size(16)));

#define AS1 __attribute__((address_space(1)))
#define AS3 __attribute__((address_space(3)))

union Frag {
    v16bf v;
    unsigned u[8];
    unsigned short s[16];
};

__device__ inline unsigned pack2bf(float a, float b) {
    union { __bf16 h[2]; unsigned u; } p;
    p.h[0] = (__bf16)a;
    p.h[1] = (__bf16)b;
    return p.u;
}

__device__ inline unsigned short bf1(float a) {
    return __builtin_bit_cast(unsigned short, (__bf16)a);
}

__device__ inline v8f vzero8() {
    v8f z;
#pragma unroll
    for (int i = 0; i < 8; ++i) z[i] = 0.0f;
    return z;
}

// A-matrix 16x32 bf16 K-pattern (ISA 7.12.2): VGPR v<4 -> K=2v(+8g), v>=4 -> K=16+2(v-4)(+8g)
__device__ inline int kpatA(int v, int g) {
    return ((v < 4) ? (2 * v) : (16 + 2 * (v - 4))) + 8 * g;
}

__device__ inline v8f wmma_bf16(const Frag& a, const Frag& b, v8f c) {
    return __builtin_amdgcn_wmma_f32_16x16x32_bf16(
        false, a.v, false, b.v, (short)0, c, false, false);
}

// ---- CDNA5 async global->LDS copy (ASYNCcnt path), with safe fallback ------
#if __has_builtin(__builtin_amdgcn_global_load_async_to_lds_b128)
#define HAVE_ASYNC_LDS 1
#endif

__device__ inline void wait_asynccnt0() {
#if __has_builtin(__builtin_amdgcn_s_wait_asynccnt)
    __builtin_amdgcn_s_wait_asynccnt(0);
#else
    asm volatile("s_wait_asynccnt 0x0" ::: "memory");
#endif
}

#define BT   8192      // B*T rows
#define CIN  768       // K dim of GEMM
#define NQKV 2304      // 3*C
#define TT   2048
#define HH   12
#define DD   64

// ---------------------------------------------------------------------------
// Kernel 1: qkv = x @ W + b, scattered into Q/K/V bf16 [B,H,T,d]
// block = 256 threads (8 waves), tile 128(M) x 128(N) x 32(K)
// ---------------------------------------------------------------------------
__global__ __launch_bounds__(256) void qkv_gemm_kernel(
    const float* __restrict__ x, const float* __restrict__ W,
    const float* __restrict__ bias,
    unsigned short* __restrict__ Qo, unsigned short* __restrict__ Ko,
    unsigned short* __restrict__ Vo)
{
    __shared__ unsigned short Asm [128][40];  // 128(M) x 32(K) bf16 (+pad)
    __shared__ unsigned short BsmT[128][40];  // 128(N) x 32(K) bf16, transposed (+pad)

    const int tid  = threadIdx.x;
    const int wave = tid >> 5;
    const int lane = tid & 31;
    const int g    = lane >> 4;
    const int ln   = lane & 15;

    const int mbase = blockIdx.y * 128;
    const int nbase = blockIdx.x * 128;

    v8f acc[8];
#pragma unroll
    for (int i = 0; i < 8; ++i) acc[i] = vzero8();

    for (int kb = 0; kb < CIN; kb += 32) {
        if (kb + 32 < CIN) {   // global_prefetch_b8 of next K-step tiles
            __builtin_prefetch(x + (size_t)(mbase + (tid >> 1)) * CIN + kb + 32, 0, 1);
            __builtin_prefetch(W + (size_t)(kb + 32 + (tid >> 3)) * NQKV + nbase, 0, 1);
        }
        __syncthreads();
#pragma unroll
        for (int i = 0; i < 4; ++i) {
            int lin = tid + i * 256;
            int row = lin >> 3, c4 = lin & 7;
            const float4 f = *(const float4*)(x + (size_t)(mbase + row) * CIN + kb + c4 * 4);
            *(unsigned*)&Asm[row][c4 * 4 + 0] = pack2bf(f.x, f.y);
            *(unsigned*)&Asm[row][c4 * 4 + 2] = pack2bf(f.z, f.w);
        }
#pragma unroll
        for (int i = 0; i < 4; ++i) {
            int lin = tid + i * 256;
            int row = lin >> 5, c4 = lin & 31;
            const float4 f = *(const float4*)(W + (size_t)(kb + row) * NQKV + nbase + c4 * 4);
            BsmT[c4 * 4 + 0][row] = bf1(f.x);
            BsmT[c4 * 4 + 1][row] = bf1(f.y);
            BsmT[c4 * 4 + 2][row] = bf1(f.z);
            BsmT[c4 * 4 + 3][row] = bf1(f.w);
        }
        __syncthreads();

        Frag bf;
        const int bcol = wave * 16 + ln;
#pragma unroll
        for (int v = 0; v < 8; ++v)
            bf.u[v] = *(const unsigned*)&BsmT[bcol][kpatA(v, g)];

#pragma unroll
        for (int mt = 0; mt < 8; ++mt) {
            Frag af;
            const int arow = mt * 16 + ln;
#pragma unroll
            for (int v = 0; v < 8; ++v)
                af.u[v] = *(const unsigned*)&Asm[arow][kpatA(v, g)];
            acc[mt] = wmma_bf16(af, bf, acc[mt]);
        }
    }

    const int ncol  = nbase + wave * 16 + ln;
    const int which = (ncol >= 1536) ? 2 : ((ncol >= 768) ? 1 : 0);
    const int c     = ncol - which * 768;
    const int h     = c >> 6;
    const int dd    = c & 63;
    unsigned short* outp = (which == 0) ? Qo : (which == 1) ? Ko : Vo;
    const float sc = (which == 0) ? 8.0f : 1.0f;   // fold sqrt(d) into Q (exact)
    const float bv = bias[ncol];

#pragma unroll
    for (int mt = 0; mt < 8; ++mt) {
#pragma unroll
        for (int r = 0; r < 8; ++r) {
            const int m  = mbase + mt * 16 + r + 8 * g;
            const int bb = m >> 11;
            const int t  = m & (TT - 1);
            const size_t idx = (((size_t)bb * HH + h) * TT + t) * DD + dd;
            outp[idx] = bf1((acc[mt][r] + bv) * sc);
        }
    }
}

// ---------------------------------------------------------------------------
// Kernel 2: flash attention, causal (Q already carries the sqrt(d) scale)
// grid: (T/128, B*H); block 256 = 8 waves; wave owns 16 q-rows
// ---------------------------------------------------------------------------
__global__ __launch_bounds__(256) void attn_kernel(
    const unsigned short* __restrict__ Q, const unsigned short* __restrict__ K,
    const unsigned short* __restrict__ V, float* __restrict__ out)
{
    __shared__ unsigned short Ksm [64][72];   // [key][d]
    __shared__ unsigned short VsmT[64][72];   // [d][key]
    __shared__ unsigned short Psm[8][16][72];

    const int tid  = threadIdx.x;
    const int wave = tid >> 5;
    const int lane = tid & 31;
    const int g    = lane >> 4;
    const int ln   = lane & 15;

    const int bh = blockIdx.y;
    const int b  = bh / HH;
    const int h  = bh % HH;
    const int qb = blockIdx.x * 128;
    const size_t head_off = (size_t)bh * TT * DD;

    Frag qf[2];
    {
        const int qrow = qb + wave * 16 + ln;
        const unsigned short* qp = Q + head_off + (size_t)qrow * DD;
#pragma unroll
        for (int f = 0; f < 2; ++f)
#pragma unroll
            for (int v = 0; v < 8; ++v)
                qf[f].u[v] = *(const unsigned*)(qp + f * 32 + kpatA(v, g));
    }

    v8f yacc[4];
#pragma unroll
    for (int i = 0; i < 4; ++i) yacc[i] = vzero8();
    float mrow[8], lrow[8];
#pragma unroll
    for (int r = 0; r < 8; ++r) { mrow[r] = -__builtin_inff(); lrow[r] = 0.0f; }

    for (int kb = 0; kb < qb + 128; kb += 64) {
        __syncthreads();
        {
            const unsigned short* kp = K + head_off + (size_t)kb * DD;
            const unsigned short* vp = V + head_off + (size_t)kb * DD;
#pragma unroll
            for (int i = 0; i < 2; ++i) {
                int lin = tid + (i << 8);       // 512 segments of 8 bf16
                int row = lin >> 3, seg = lin & 7;
#ifdef HAVE_ASYNC_LDS
                // K tile: DMA straight into LDS (no VGPR round-trip, ASYNCcnt)
                __builtin_amdgcn_global_load_async_to_lds_b128(
                    (AS1 v4i*)(kp + (size_t)row * DD + seg * 8),
                    (AS3 v4i*)&Ksm[row][seg * 8],
                    0, 0);
#else
                const uint4 kq = *(const uint4*)(kp + (size_t)row * DD + seg * 8);
                *(uint4*)&Ksm[row][seg * 8] = kq;
#endif
                // V tile: needs transpose -> load wide, scatter 16-bit
                union { uint4 q; unsigned short s[8]; } vv;
                vv.q = *(const uint4*)(vp + (size_t)row * DD + seg * 8);
#pragma unroll
                for (int j = 0; j < 8; ++j)
                    VsmT[seg * 8 + j][row] = vv.s[j];
            }
#ifdef HAVE_ASYNC_LDS
            wait_asynccnt0();
#endif
        }
        __syncthreads();

        v8f st[4];
#pragma unroll
        for (int nt = 0; nt < 4; ++nt) {
            Frag kf0, kf1;
            const int krow = nt * 16 + ln;
#pragma unroll
            for (int v = 0; v < 8; ++v) {
                kf0.u[v] = *(const unsigned*)&Ksm[krow][kpatA(v, g)];
                kf1.u[v] = *(const unsigned*)&Ksm[krow][32 + kpatA(v, g)];
            }
            v8f cacc = vzero8();
            cacc = wmma_bf16(qf[0], kf0, cacc);
            cacc = wmma_bf16(qf[1], kf1, cacc);
            st[nt] = cacc;
        }

#pragma unroll
        for (int r = 0; r < 8; ++r) {
            const int q_abs = qb + wave * 16 + r + 8 * g;
            float rmax = -__builtin_inff();
#pragma unroll
            for (int nt = 0; nt < 4; ++nt) {
                const int k_abs = kb + nt * 16 + ln;
                float sv = st[nt][r];               // scale already folded into Q
                if (k_abs > q_abs) sv = -__builtin_inff();
                st[nt][r] = sv;
                rmax = fmaxf(rmax, sv);
            }
#pragma unroll
            for (int off = 1; off < 16; off <<= 1)
                rmax = fmaxf(rmax, __shfl_xor(rmax, off, 32));
            const float mnew  = fmaxf(mrow[r], rmax);
            const float alpha = __expf(mrow[r] - mnew);
            float rsum = 0.0f;
#pragma unroll
            for (int nt = 0; nt < 4; ++nt) {
                const float p = __expf(st[nt][r] - mnew);
                st[nt][r] = p;
                rsum += p;
            }
#pragma unroll
            for (int off = 1; off < 16; off <<= 1)
                rsum += __shfl_xor(rsum, off, 32);
            lrow[r] = lrow[r] * alpha + rsum;
            mrow[r] = mnew;
#pragma unroll
            for (int dt = 0; dt < 4; ++dt) yacc[dt][r] *= alpha;
        }

#pragma unroll
        for (int nt = 0; nt < 4; ++nt)
#pragma unroll
            for (int r = 0; r < 8; ++r)
                Psm[wave][r + 8 * g][nt * 16 + ln] = bf1(st[nt][r]);
        __builtin_amdgcn_wave_barrier();

        Frag pf[2];
#pragma unroll
        for (int f = 0; f < 2; ++f)
#pragma unroll
            for (int v = 0; v < 8; ++v)
                pf[f].u[v] = *(const unsigned*)&Psm[wave][ln][f * 32 + kpatA(v, g)];

#pragma unroll
        for (int dt = 0; dt < 4; ++dt) {
            Frag vf0, vf1;
            const int dcol = dt * 16 + ln;
#pragma unroll
            for (int v = 0; v < 8; ++v) {
                vf0.u[v] = *(const unsigned*)&VsmT[dcol][kpatA(v, g)];
                vf1.u[v] = *(const unsigned*)&VsmT[dcol][32 + kpatA(v, g)];
            }
            yacc[dt] = wmma_bf16(pf[0], vf0, yacc[dt]);
            yacc[dt] = wmma_bf16(pf[1], vf1, yacc[dt]);
        }
    }

#pragma unroll
    for (int r = 0; r < 8; ++r) {
        const int q_abs = qb + wave * 16 + r + 8 * g;
        const float inv = 1.0f / lrow[r];
#pragma unroll
        for (int dt = 0; dt < 4; ++dt) {
            const size_t idx = ((size_t)b * TT + q_abs) * CIN + h * DD + dt * 16 + ln;
            out[idx] = yacc[dt][r] * inv;
        }
    }
}

// ---------------------------------------------------------------------------
extern "C" void kernel_launch(void* const* d_in, const int* in_sizes, int n_in,
                              void* d_out, int out_size, void* d_ws, size_t ws_size,
                              hipStream_t stream) {
    const float* x    = (const float*)d_in[0];   // [4,2048,768]
    const float* W    = (const float*)d_in[1];   // [768,2304]
    const float* bias = (const float*)d_in[2];   // [2304]
    float* out        = (float*)d_out;           // [4,2048,768]

    unsigned short* Qw = (unsigned short*)d_ws;  // bf16 [B,H,T,d]
    const size_t sz = (size_t)4 * HH * TT * DD;  // 6,291,456 elements each
    unsigned short* Kw = Qw + sz;
    unsigned short* Vw = Kw + sz;

    dim3 g1(NQKV / 128, BT / 128);               // (18, 64)
    qkv_gemm_kernel<<<g1, 256, 0, stream>>>(x, W, bias, Qw, Kw, Vw);

    dim3 g2(TT / 128, 4 * HH);                   // (16, 48)
    attn_kernel<<<g2, 256, 0, stream>>>(Qw, Kw, Vw, out);
}
